// get_model_55336358642811
// MI455X (gfx1250) — matrix operand order repaired and checked by reading, not compiled, not run on hardware
//
#include <hip/hip_runtime.h>
#include <hip/hip_bf16.h>

// ---------------------------------------------------------------------------
// PointNet++ (SSG) classification for MI455X / gfx1250.
// GEMMs: v_wmma_f32_16x16x32_f16 with pre-packed, zero-padded f16 weights and
// bias+BN folded to per-channel (scale, shift). Async-to-LDS staging in the
// final max reduction.
// ---------------------------------------------------------------------------

typedef __attribute__((ext_vector_type(16))) _Float16 v16h;
typedef __attribute__((ext_vector_type(8)))  _Float16 v8h;
typedef __attribute__((ext_vector_type(8)))  float    v8f;

#define BN_EPS 1e-5f
#define NBATCH 16
#define NPTS   4096

// ----------------------- WMMA fragment loaders -----------------------------
// A fragment 16x32 f16 from LDS act [rows][LDA] (row-major in K).
// lane L: row M=(L&15); halves 0..7 = K=k0+8*(L>>4)+i; halves 8..15 = +16.
template <int LDA>
__device__ __forceinline__ v16h load_a(const _Float16* act, int m0, int k0,
                                       int lane) {
    const int row  = m0 + (lane & 15);
    const int ksel = lane >> 4;
    const v8h* p = (const v8h*)(act + row * LDA + k0 + 8 * ksel);
    v8h lo = p[0];   // K offsets +0..7
    v8h hi = p[2];   // K offsets +16..23
    return __builtin_shufflevector(lo, hi, 0, 1, 2, 3, 4, 5, 6, 7, 8, 9, 10,
                                   11, 12, 13, 14, 15);
}

// B fragment 32x16 f16 from pre-packed global Wh[loutp][LIN] (f16, padded).
// lane L: col N=(L&15); halves 0..7 = K=k0+8*(L>>4)+i; halves 8..15 = +16.
template <int LIN>
__device__ __forceinline__ v16h load_b(const _Float16* __restrict__ wh, int n0,
                                       int k0, int lane) {
    const int n  = n0 + (lane & 15);
    const int kb = k0 + 8 * (lane >> 4);
    const v8h* p = (const v8h*)(wh + (size_t)n * LIN + kb);
    v8h lo = p[0];
    v8h hi = p[2];
    return __builtin_shufflevector(lo, hi, 0, 1, 2, 3, 4, 5, 6, 7, 8, 9, 10,
                                   11, 12, 13, 14, 15);
}

// ----------------------- fused GEMM + (scale,shift) + ReLU -----------------
// actIn: LDS f16 [ROWS][LIN]; actOut: LDS f16 [ROWS][LOUT].
// wh: [LOUT][LIN] f16 zero-padded; ab: [0..LOUT)=scale, [LOUT..2*LOUT)=shift.
template <int ROWS, int LIN, int LOUT, bool RELU>
__device__ __forceinline__ void mlp_layer_t(const _Float16* actIn,
                                            _Float16* actOut,
                                            const _Float16* __restrict__ wh,
                                            const float* __restrict__ ab) {
    const int lane   = threadIdx.x & 31;
    const int wave   = threadIdx.x >> 5;
    const int nwaves = blockDim.x >> 5;
    constexpr int MT = ROWS / 16;
    constexpr int NT = LOUT / 16;
    for (int t = wave; t < MT * NT; t += nwaves) {
        const int m0 = (t / NT) * 16;
        const int n0 = (t % NT) * 16;
        __builtin_prefetch(wh + (size_t)n0 * LIN, 0, 1);
        v8f c = {};
#pragma unroll
        for (int k0 = 0; k0 < LIN; k0 += 32) {
            v16h a = load_a<LIN>(actIn, m0, k0, lane);
            v16h b = load_b<LIN>(wh, n0, k0, lane);
            c = __builtin_amdgcn_wmma_f32_16x16x32_f16(false, a, false, b,
                                                       (short)0, c, false,
                                                       false);
        }
        const int n     = n0 + (lane & 15);
        const int mh    = lane >> 4;
        const float scl = ab[n];
        const float sft = ab[LOUT + n];
#pragma unroll
        for (int i = 0; i < 8; ++i) {
            const int m = m0 + i + 8 * mh;
            float v = fmaf(scl, c[i], sft);
            if (RELU) v = fmaxf(v, 0.0f);
            actOut[m * LOUT + n] = (_Float16)v;
        }
    }
}

// Final FC layer: f32 output to global, no ReLU, guard n < COUT.
template <int ROWS, int LIN, int LOUT, int COUT>
__device__ __forceinline__ void mlp_layer_gout_t(const _Float16* actIn,
                                                 const _Float16* __restrict__ wh,
                                                 const float* __restrict__ ab,
                                                 float* __restrict__ gout,
                                                 int ldg) {
    const int lane   = threadIdx.x & 31;
    const int wave   = threadIdx.x >> 5;
    const int nwaves = blockDim.x >> 5;
    constexpr int MT = ROWS / 16;
    constexpr int NT = LOUT / 16;
    for (int t = wave; t < MT * NT; t += nwaves) {
        const int m0 = (t / NT) * 16;
        const int n0 = (t % NT) * 16;
        v8f c = {};
#pragma unroll
        for (int k0 = 0; k0 < LIN; k0 += 32) {
            v16h a = load_a<LIN>(actIn, m0, k0, lane);
            v16h b = load_b<LIN>(wh, n0, k0, lane);
            c = __builtin_amdgcn_wmma_f32_16x16x32_f16(false, a, false, b,
                                                       (short)0, c, false,
                                                       false);
        }
        const int n = n0 + (lane & 15);
        if (n < COUT) {
            const int mh    = lane >> 4;
            const float scl = ab[n];
            const float sft = ab[LOUT + n];
#pragma unroll
            for (int i = 0; i < 8; ++i) {
                const int m = m0 + i + 8 * mh;
                gout[(size_t)m * ldg + n] = fmaf(scl, c[i], sft);
            }
        }
    }
}

// ---------------------------- weight pre-pack ------------------------------
// W f32 [cout][cin] -> Wh f16 [loutp][linp] zero-padded; fold bias + BN into
// per-channel scale/shift: y = scale*acc + shift (shift = scale*b + beta - mu*scale).
__global__ void prep_weights_kernel(const float* __restrict__ w,
                                    const float* __restrict__ b,
                                    const float* __restrict__ g,
                                    const float* __restrict__ be,
                                    const float* __restrict__ mu,
                                    const float* __restrict__ va, int cin,
                                    int cout, int linp, int loutp,
                                    _Float16* __restrict__ wh,
                                    float* __restrict__ ab) {
    int tid    = blockIdx.x * blockDim.x + threadIdx.x;
    int stride = gridDim.x * blockDim.x;
    int total  = linp * loutp;
    for (int idx = tid; idx < total; idx += stride) {
        int n = idx / linp, k = idx % linp;
        float v = (n < cout && k < cin) ? w[(size_t)n * cin + k] : 0.0f;
        wh[idx] = (_Float16)v;
    }
    for (int n = tid; n < loutp; n += stride) {
        float scale = 0.0f, shift = 0.0f;
        if (n < cout) {
            if (g) {
                scale = g[n] * rsqrtf(va[n] + BN_EPS);
                shift = fmaf(scale, b[n], be[n] - mu[n] * scale);
            } else {
                scale = 1.0f;
                shift = b[n];
            }
        }
        ab[n]         = scale;
        ab[loutp + n] = shift;
    }
}

// ---------------------------- transpose (B,3,N)->(B,N,3) -------------------
__global__ void transpose_kernel(const float* __restrict__ pts,
                                 float* __restrict__ xyz, int total) {
    int i = blockIdx.x * blockDim.x + threadIdx.x;
    if (i >= total) return;
    int n = i % NPTS;
    int c = (i / NPTS) % 3;
    int b = i / (3 * NPTS);
    xyz[((size_t)b * NPTS + n) * 3 + c] = pts[i];
}

// ---------------------------- farthest point sampling ----------------------
__global__ void fps_kernel(const float* __restrict__ xyz, int Nn, int npoint,
                           int* __restrict__ out) {
    int b = blockIdx.x;
    __shared__ float dist[4096];
    __shared__ float rv[256];
    __shared__ int   ri[256];
    __shared__ int   sfar;
    const float* p = xyz + (size_t)b * Nn * 3;
    for (int n = threadIdx.x; n < Nn; n += blockDim.x) dist[n] = 1e10f;
    if (threadIdx.x == 0) sfar = 0;
    __syncthreads();
    for (int it = 0; it < npoint; ++it) {
        int far = sfar;
        if (threadIdx.x == 0) out[(size_t)b * npoint + it] = far;
        float cx = p[far * 3 + 0], cy = p[far * 3 + 1], cz = p[far * 3 + 2];
        float bestv = -1.0f;
        int   besti = 0x7fffffff;
        for (int n = threadIdx.x; n < Nn; n += blockDim.x) {
            float dx = p[n * 3 + 0] - cx;
            float dy = p[n * 3 + 1] - cy;
            float dz = p[n * 3 + 2] - cz;
            float d  = dx * dx + dy * dy + dz * dz;
            float dn = fminf(dist[n], d);
            dist[n]  = dn;
            if (dn > bestv) { bestv = dn; besti = n; }  // first-index ties
        }
        rv[threadIdx.x] = bestv;
        ri[threadIdx.x] = besti;
        __syncthreads();
        for (int off = blockDim.x >> 1; off > 0; off >>= 1) {
            if (threadIdx.x < off) {
                float v2 = rv[threadIdx.x + off];
                int   i2 = ri[threadIdx.x + off];
                if (v2 > rv[threadIdx.x] ||
                    (v2 == rv[threadIdx.x] && i2 < ri[threadIdx.x])) {
                    rv[threadIdx.x] = v2;
                    ri[threadIdx.x] = i2;
                }
            }
            __syncthreads();
        }
        if (threadIdx.x == 0) sfar = ri[0];
        __syncthreads();
    }
}

// ---------------------------- gather sampled centroids ---------------------
__global__ void gather_kernel(const float* __restrict__ xyz, int Nn,
                              const int* __restrict__ idx, int S,
                              float* __restrict__ out) {
    int t = blockIdx.x * blockDim.x + threadIdx.x;
    if (t >= NBATCH * S) return;
    int b = t / S;
    int j = idx[t];
    const float* p = xyz + ((size_t)b * Nn + j) * 3;
    out[(size_t)t * 3 + 0] = p[0];
    out[(size_t)t * 3 + 1] = p[1];
    out[(size_t)t * 3 + 2] = p[2];
}

// ---------------------------- ball query -----------------------------------
__global__ void ballq_kernel(const float* __restrict__ xyz, int Nn,
                             const float* __restrict__ nx, int S, float r2,
                             int ns, int* __restrict__ gidx) {
    int t = blockIdx.x * blockDim.x + threadIdx.x;
    if (t >= NBATCH * S) return;
    int b = t / S;
    const float* p = xyz + (size_t)b * Nn * 3;
    float cx = nx[(size_t)t * 3 + 0];
    float cy = nx[(size_t)t * 3 + 1];
    float cz = nx[(size_t)t * 3 + 2];
    int* out  = gidx + (size_t)t * ns;
    int found = 0, first = -1;
    for (int n = 0; n < Nn && found < ns; ++n) {
        float dx = p[n * 3 + 0] - cx;
        float dy = p[n * 3 + 1] - cy;
        float dz = p[n * 3 + 2] - cz;
        if (dx * dx + dy * dy + dz * dz <= r2) {
            if (first < 0) first = n;
            out[found++] = n;
        }
    }
    if (first < 0) first = 0;
    for (; found < ns; ++found) out[found] = first;
}

// ---------------------------- SA stage 1 -----------------------------------
__global__ void sa1_kernel(const float* __restrict__ xyz,
                           const float* __restrict__ newxyz,
                           const int* __restrict__ gidx,
                           const _Float16* __restrict__ w0,
                           const float* __restrict__ a0,
                           const _Float16* __restrict__ w1,
                           const float* __restrict__ a1,
                           const _Float16* __restrict__ w2,
                           const float* __restrict__ a2,
                           float* __restrict__ outp) {
    int bs = blockIdx.x;  // b*512 + s
    int b  = bs >> 9;
    __shared__ _Float16 buf0[32 * 64];
    __shared__ _Float16 buf1[32 * 128];
    const int* gi = gidx + (size_t)bs * 32;
    float cx = newxyz[(size_t)bs * 3 + 0];
    float cy = newxyz[(size_t)bs * 3 + 1];
    float cz = newxyz[(size_t)bs * 3 + 2];
    for (int i = threadIdx.x; i < 32 * 32; i += blockDim.x)
        buf0[i] = (_Float16)0.0f;
    __syncthreads();
    for (int r = threadIdx.x; r < 32; r += blockDim.x) {
        int j = gi[r];
        const float* p = xyz + ((size_t)b * NPTS + j) * 3;
        buf0[r * 32 + 0] = (_Float16)(p[0] - cx);
        buf0[r * 32 + 1] = (_Float16)(p[1] - cy);
        buf0[r * 32 + 2] = (_Float16)(p[2] - cz);
    }
    __syncthreads();
    mlp_layer_t<32, 32, 64, true>(buf0, buf1, w0, a0);   __syncthreads();
    mlp_layer_t<32, 64, 64, true>(buf1, buf0, w1, a1);   __syncthreads();
    mlp_layer_t<32, 64, 128, true>(buf0, buf1, w2, a2);  __syncthreads();
    float* out = outp + (size_t)bs * 128;
    for (int n = threadIdx.x; n < 128; n += blockDim.x) {
        float mx = -1e30f;
#pragma unroll 8
        for (int r = 0; r < 32; ++r)
            mx = fmaxf(mx, (float)buf1[r * 128 + n]);
        out[n] = mx;
    }
}

// ---------------------------- SA stage 2 -----------------------------------
__global__ void sa2_kernel(const float* __restrict__ l1x,
                           const float* __restrict__ l1p,
                           const float* __restrict__ newxyz,
                           const int* __restrict__ gidx,
                           const _Float16* __restrict__ w0,
                           const float* __restrict__ a0,
                           const _Float16* __restrict__ w1,
                           const float* __restrict__ a1,
                           const _Float16* __restrict__ w2,
                           const float* __restrict__ a2,
                           float* __restrict__ outp) {
    int bs = blockIdx.x;  // b*128 + s
    int b  = bs >> 7;
    __shared__ _Float16 buf0[64 * 160];
    __shared__ _Float16 buf1[64 * 256];
    const int* gi = gidx + (size_t)bs * 64;
    float cx = newxyz[(size_t)bs * 3 + 0];
    float cy = newxyz[(size_t)bs * 3 + 1];
    float cz = newxyz[(size_t)bs * 3 + 2];
    for (int i = threadIdx.x; i < 64 * 160; i += blockDim.x)
        buf0[i] = (_Float16)0.0f;
    __syncthreads();
    for (int e = threadIdx.x; e < 64 * 131; e += blockDim.x) {
        int r = e / 131, c = e % 131;
        int j = gi[r];
        float v;
        if (c == 0)      v = l1x[((size_t)b * 512 + j) * 3 + 0] - cx;
        else if (c == 1) v = l1x[((size_t)b * 512 + j) * 3 + 1] - cy;
        else if (c == 2) v = l1x[((size_t)b * 512 + j) * 3 + 2] - cz;
        else             v = l1p[((size_t)b * 512 + j) * 128 + (c - 3)];
        buf0[r * 160 + c] = (_Float16)v;
    }
    __syncthreads();
    mlp_layer_t<64, 160, 128, true>(buf0, buf1, w0, a0);  __syncthreads();
    mlp_layer_t<64, 128, 128, true>(buf1, buf0, w1, a1);  __syncthreads();
    mlp_layer_t<64, 128, 256, true>(buf0, buf1, w2, a2);  __syncthreads();
    float* out = outp + (size_t)bs * 256;
    for (int n = threadIdx.x; n < 256; n += blockDim.x) {
        float mx = -1e30f;
#pragma unroll 8
        for (int r = 0; r < 64; ++r)
            mx = fmaxf(mx, (float)buf1[r * 256 + n]);
        out[n] = mx;
    }
}

// ---------------------------- SA stage 3 (group_all, chunked) --------------
__global__ void sa3_kernel(const float* __restrict__ l2x,
                           const float* __restrict__ l2p,
                           const _Float16* __restrict__ w0,
                           const float* __restrict__ a0,
                           const _Float16* __restrict__ w1,
                           const float* __restrict__ a1,
                           const _Float16* __restrict__ w2,
                           const float* __restrict__ a2,
                           float* __restrict__ partial) {
    int bc = blockIdx.x;  // b*8 + chunk
    int b = bc >> 3, chunk = bc & 7;
    __shared__ _Float16 buf0[16 * 512];
    __shared__ _Float16 buf1[16 * 1024];
    for (int i = threadIdx.x; i < 16 * 288; i += blockDim.x)
        buf0[i] = (_Float16)0.0f;
    __syncthreads();
    for (int e = threadIdx.x; e < 16 * 259; e += blockDim.x) {
        int r = e / 259, c = e % 259;
        int rg = chunk * 16 + r;
        float v;
        if (c < 3) v = l2x[((size_t)b * 128 + rg) * 3 + c];
        else       v = l2p[((size_t)b * 128 + rg) * 256 + (c - 3)];
        buf0[r * 288 + c] = (_Float16)v;
    }
    __syncthreads();
    mlp_layer_t<16, 288, 256, true>(buf0, buf1, w0, a0);   __syncthreads();
    mlp_layer_t<16, 256, 512, true>(buf1, buf0, w1, a1);   __syncthreads();
    mlp_layer_t<16, 512, 1024, true>(buf0, buf1, w2, a2);  __syncthreads();
    float* out = partial + (size_t)bc * 1024;
    for (int n = threadIdx.x; n < 1024; n += blockDim.x) {
        float mx = -1e30f;
#pragma unroll
        for (int r = 0; r < 16; ++r)
            mx = fmaxf(mx, (float)buf1[r * 1024 + n]);
        out[n] = mx;
    }
}

// Reduce 8 chunk-partials -> l3p; stage via async global->LDS copy (ASYNCcnt).
__global__ void l3red_kernel(const float* __restrict__ partial,
                             float* __restrict__ l3p,
                             float* __restrict__ l3out) {
    int b = blockIdx.x;
    __shared__ float sbuf[8 * 1024];  // 32 KB
    const float* src = partial + (size_t)b * 8192;
    // Each lane async-copies 16B chunks directly into LDS (no VGPR data path).
    for (int i = threadIdx.x * 4; i < 8192; i += blockDim.x * 4) {
        unsigned           ldsoff = (unsigned)(uintptr_t)(&sbuf[i]);
        unsigned long long gaddr  = (unsigned long long)(uintptr_t)(src + i);
        asm volatile("global_load_async_to_lds_b128 %0, %1, off"
                     :
                     : "v"(ldsoff), "v"(gaddr)
                     : "memory");
    }
    asm volatile("s_wait_asynccnt 0x0" ::: "memory");
    __syncthreads();
    for (int n = threadIdx.x; n < 1024; n += blockDim.x) {
        float mx = -1e30f;
#pragma unroll
        for (int k = 0; k < 8; ++k) mx = fmaxf(mx, sbuf[k * 1024 + n]);
        l3p[(size_t)b * 1024 + n]   = mx;
        l3out[(size_t)b * 1024 + n] = mx;  // (B,1024,1) flat == (B,1024)
    }
}

// ---------------------------- FC head --------------------------------------
__global__ void fc_kernel(const float* __restrict__ l3p,
                          const _Float16* __restrict__ w0,
                          const float* __restrict__ a0,
                          const _Float16* __restrict__ w1,
                          const float* __restrict__ a1,
                          const _Float16* __restrict__ w2,
                          const float* __restrict__ a2,
                          float* __restrict__ logits) {
    __shared__ _Float16 buf0[16 * 1024];
    __shared__ _Float16 buf1[16 * 512];
    for (int e = threadIdx.x; e < 16 * 1024; e += blockDim.x)
        buf0[e] = (_Float16)l3p[e];
    __syncthreads();
    mlp_layer_t<16, 1024, 512, true>(buf0, buf1, w0, a0);  __syncthreads();
    mlp_layer_t<16, 512, 256, true>(buf1, buf0, w1, a1);   __syncthreads();
    mlp_layer_gout_t<16, 256, 48, 40>(buf0, w2, a2, logits, 40);
}

// ---------------------------------------------------------------------------
extern "C" void kernel_launch(void* const* d_in, const int* in_sizes, int n_in,
                              void* d_out, int out_size, void* d_ws,
                              size_t ws_size, hipStream_t stream) {
    (void)in_sizes; (void)n_in; (void)out_size; (void)ws_size;
    const int B = NBATCH, N = NPTS;
    const float* pts = (const float*)d_in[0];

    // Layer table: d_in base, cin, cout, padded K (linp), padded N (loutp), bn
    struct Desc { int base, cin, cout, linp, loutp, bn; };
    const Desc ds[12] = {
        {1, 3, 64, 32, 64, 1},       {7, 64, 64, 64, 64, 1},
        {13, 64, 128, 64, 128, 1},   {19, 131, 128, 160, 128, 1},
        {25, 128, 128, 128, 128, 1}, {31, 128, 256, 128, 256, 1},
        {37, 259, 256, 288, 256, 1}, {43, 256, 512, 256, 512, 1},
        {49, 512, 1024, 512, 1024, 1}, {55, 1024, 512, 1024, 512, 1},
        {61, 512, 256, 512, 256, 1}, {67, 256, 40, 256, 48, 0},
    };

    // ---- workspace carve: packed f16 weights first (aligned), then scale/shift tables,
    // then float buffers, then ints.
    _Float16* hbase = (_Float16*)d_ws;
    _Float16* whp[12];
    size_t hoff = 0;
    for (int i = 0; i < 12; ++i) {
        whp[i] = hbase + hoff;
        hoff += (size_t)ds[i].linp * ds[i].loutp;
    }
    hoff = (hoff + 63) & ~(size_t)63;

    float* f = (float*)(hbase + hoff);
    float* abp[12];
    for (int i = 0; i < 12; ++i) {
        abp[i] = f;
        f += 2 * ds[i].loutp;
    }
    float* xyz    = f;  f += (size_t)B * N * 3;
    float* l1x    = f;  f += (size_t)B * 512 * 3;
    float* l1p    = f;  f += (size_t)B * 512 * 128;
    float* l2x    = f;  f += (size_t)B * 128 * 3;
    float* l2p    = f;  f += (size_t)B * 128 * 256;
    float* l3part = f;  f += (size_t)B * 8 * 1024;
    float* l3p    = f;  f += (size_t)B * 1024;
    int* ip    = (int*)f;
    int* fps1  = ip;  ip += B * 512;
    int* fps2  = ip;  ip += B * 128;
    int* gidx1 = ip;  ip += B * 512 * 32;
    int* gidx2 = ip;  ip += B * 128 * 64;

    float* logits = (float*)d_out;    // (16,40)
    float* l3out  = logits + B * 40;  // (16,1024,1)

    // ---- pre-pack all 12 layers ----
    for (int i = 0; i < 12; ++i) {
        const Desc& d = ds[i];
        const float* w  = (const float*)d_in[d.base];
        const float* bb = (const float*)d_in[d.base + 1];
        const float *g = nullptr, *be = nullptr, *mu = nullptr, *va = nullptr;
        if (d.bn) {
            g  = (const float*)d_in[d.base + 2];
            be = (const float*)d_in[d.base + 3];
            mu = (const float*)d_in[d.base + 4];
            va = (const float*)d_in[d.base + 5];
        }
        int total  = d.linp * d.loutp;
        int blocks = (total + 255) / 256;
        if (blocks > 512) blocks = 512;
        prep_weights_kernel<<<blocks, 256, 0, stream>>>(
            w, bb, g, be, mu, va, d.cin, d.cout, d.linp, d.loutp, whp[i],
            abp[i]);
    }

    transpose_kernel<<<(B * 3 * N + 255) / 256, 256, 0, stream>>>(pts, xyz,
                                                                  B * 3 * N);
    // ---- SA1 ----
    fps_kernel<<<B, 256, 0, stream>>>(xyz, N, 512, fps1);
    gather_kernel<<<(B * 512 + 255) / 256, 256, 0, stream>>>(xyz, N, fps1, 512,
                                                             l1x);
    ballq_kernel<<<(B * 512 + 255) / 256, 256, 0, stream>>>(
        xyz, N, l1x, 512, 0.2f * 0.2f, 32, gidx1);
    sa1_kernel<<<B * 512, 128, 0, stream>>>(xyz, l1x, gidx1, whp[0], abp[0],
                                            whp[1], abp[1], whp[2], abp[2],
                                            l1p);
    // ---- SA2 ----
    fps_kernel<<<B, 256, 0, stream>>>(l1x, 512, 128, fps2);
    gather_kernel<<<(B * 128 + 255) / 256, 256, 0, stream>>>(l1x, 512, fps2,
                                                             128, l2x);
    ballq_kernel<<<(B * 128 + 255) / 256, 256, 0, stream>>>(
        l1x, 512, l2x, 128, 0.4f * 0.4f, 64, gidx2);
    sa2_kernel<<<B * 128, 128, 0, stream>>>(l1x, l1p, l2x, gidx2, whp[3],
                                            abp[3], whp[4], abp[4], whp[5],
                                            abp[5], l2p);
    // ---- SA3 (group_all) ----
    sa3_kernel<<<B * 8, 128, 0, stream>>>(l2x, l2p, whp[6], abp[6], whp[7],
                                          abp[7], whp[8], abp[8], l3part);
    l3red_kernel<<<B, 256, 0, stream>>>(l3part, l3p, l3out);
    // ---- FC head ----
    fc_kernel<<<1, 128, 0, stream>>>(l3p, whp[9], abp[9], whp[10], abp[10],
                                     whp[11], abp[11], logits);
}
